// QuantumAttention_67508295959008
// MI455X (gfx1250) — matrix operand order repaired and checked by reading, not compiled, not run on hardware
//
#include <hip/hip_runtime.h>

// ---------------------------------------------------------------------------
// Types for CDNA5 WMMA (wave32): v_wmma_f32_16x16x32_bf16
// ---------------------------------------------------------------------------
typedef __attribute__((ext_vector_type(16))) __bf16        v16bf;
typedef __attribute__((ext_vector_type(8)))  __bf16        v8bf;
typedef __attribute__((ext_vector_type(8)))  float         v8f;
typedef __attribute__((ext_vector_type(16))) unsigned short v16u;

static __device__ __forceinline__ v8f vzero8() {
  v8f a;
#pragma unroll
  for (int i = 0; i < 8; ++i) a[i] = 0.0f;
  return a;
}

// A-operand (16x32 bf16): lane holds two 8-element (16B) segments:
//   K in [hi*8, hi*8+8) and [16+hi*8, 16+hi*8+8). p points at first segment,
//   second segment is p+16 elements.
static __device__ __forceinline__ v16bf ld_seg2(const __bf16* p) {
  v8bf lo = *(const v8bf*)p;
  v8bf hi = *(const v8bf*)(p + 16);
  return __builtin_shufflevector(lo, hi, 0, 1, 2, 3, 4, 5, 6, 7,
                                 8, 9, 10, 11, 12, 13, 14, 15);
}

// B-operand (32x16 bf16): lane holds two 8-element segments: K in
//   [hi*8, hi*8+8) and [16+hi*8, 16+hi*8+8); second segment is p+16.
static __device__ __forceinline__ v16bf ld_cont16(const __bf16* p) {
  v8bf lo = *(const v8bf*)p;
  v8bf hi = *(const v8bf*)(p + 16);
  return __builtin_shufflevector(lo, hi, 0, 1, 2, 3, 4, 5, 6, 7,
                                 8, 9, 10, 11, 12, 13, 14, 15);
}

// bf16 vector negate via sign-bit XOR (WMMA NEG bits are C-only for bf16).
static __device__ __forceinline__ v16bf negv(v16bf x) {
  v16u u;
  __builtin_memcpy(&u, &x, sizeof(u));
  u ^= (unsigned short)0x8000;
  v16bf r;
  __builtin_memcpy(&r, &u, sizeof(r));
  return r;
}

static __device__ __forceinline__ v8f wmma_bf16(v16bf a, v16bf b, v8f c) {
  // (neg_a, A, neg_b, B, c_mod, C, reuse_a, reuse_b)
  return __builtin_amdgcn_wmma_f32_16x16x32_bf16(false, a, false, b,
                                                 (short)0, c, false, false);
}

// ---------------------------------------------------------------------------
// Constants: DIM=1024, HEADS=16, DH=64, B=4, N=1024.  M = B*N = 4096.
// ---------------------------------------------------------------------------
#define KDIM   1024
#define NROW   4096           // B*N
#define LAYERW 1048576        // 1024*1024 per weight layer
#define BUFSZ  4194304        // elements per bf16 workspace buffer

// ---------------------------------------------------------------------------
// fp32 -> bf16 conversion (exact grid: 2048 blocks * 256 thr * 8 elem = 4M)
// ---------------------------------------------------------------------------
__global__ __launch_bounds__(256) void convert_f32_bf16_kernel(
    const float* __restrict__ src, __bf16* __restrict__ dst) {
  size_t i = ((size_t)blockIdx.x * 256 + threadIdx.x) * 8;
  float4 a = *(const float4*)(src + i);
  float4 b = *(const float4*)(src + i + 4);
  v8bf o;
  o[0] = (__bf16)a.x; o[1] = (__bf16)a.y; o[2] = (__bf16)a.z; o[3] = (__bf16)a.w;
  o[4] = (__bf16)b.x; o[5] = (__bf16)b.y; o[6] = (__bf16)b.z; o[7] = (__bf16)b.w;
  *(v8bf*)(dst + i) = o;
}

// ---------------------------------------------------------------------------
// Complex GEMM 16x16 tile core:  Yr = Xr Wr^T - Xi Wi^T ;  Yi = Xr Wi^T + Xi Wr^T
// X: [rows, 1024] bf16 row-major, W: [1024, 1024] bf16 row-major (used as W^T).
// ---------------------------------------------------------------------------
static __device__ __forceinline__ void complex_gemm_tile(
    const __bf16* __restrict__ Xr, const __bf16* __restrict__ Xi,
    const __bf16* __restrict__ Wr, const __bf16* __restrict__ Wi,
    int mT, int nT, int lane, v8f& accr, v8f& acci) {
  const int ln = lane & 15;
  const int hi = lane >> 4;
  const __bf16* xr_p = Xr + (size_t)(mT * 16 + ln) * KDIM + hi * 8;
  const __bf16* xi_p = Xi + (size_t)(mT * 16 + ln) * KDIM + hi * 8;
  const __bf16* wr_p = Wr + (size_t)(nT * 16 + ln) * KDIM + hi * 8;
  const __bf16* wi_p = Wi + (size_t)(nT * 16 + ln) * KDIM + hi * 8;
  accr = vzero8();
  acci = vzero8();
#pragma unroll 4
  for (int kk = 0; kk < KDIM; kk += 32) {
    __builtin_prefetch(wr_p + kk + 256, 0, 1);
    __builtin_prefetch(wi_p + kk + 256, 0, 1);
    v16bf axr = ld_seg2(xr_p + kk);
    v16bf axi = ld_seg2(xi_p + kk);
    v16bf bwr = ld_cont16(wr_p + kk);
    v16bf bwi = ld_cont16(wi_p + kk);
    accr = wmma_bf16(axr, bwr, accr);
    accr = wmma_bf16(negv(axi), bwi, accr);   // - Xi*Wi^T
    acci = wmma_bf16(axr, bwi, acci);
    acci = wmma_bf16(axi, bwr, acci);
  }
}

// ---------------------------------------------------------------------------
// QKV projection: 3 layers x 256 mTiles x 64 nTiles = 49152 wave-tiles.
// blockDim=256 (8 waves) -> 6144 blocks.  Writes bf16 head-major Q,K and
// transposed V ([B,H,dh,N]) for the PV GEMM.
// ---------------------------------------------------------------------------
__global__ __launch_bounds__(256) void proj_qkv_kernel(
    const __bf16* __restrict__ Xr, const __bf16* __restrict__ Xi,
    const __bf16* __restrict__ Wr4, const __bf16* __restrict__ Wi4,
    const float* __restrict__ br4, const float* __restrict__ bi4,
    __bf16* __restrict__ Qr, __bf16* __restrict__ Qi,
    __bf16* __restrict__ Kr, __bf16* __restrict__ Ki,
    __bf16* __restrict__ Vrt, __bf16* __restrict__ Vit) {
  const int wave = blockIdx.x * 8 + (threadIdx.x >> 5);
  const int lane = threadIdx.x & 31;
  const int layer = wave / 16384;
  const int rem = wave - layer * 16384;
  const int mT = rem >> 6;
  const int nT = rem & 63;

  v8f accr, acci;
  complex_gemm_tile(Xr, Xi, Wr4 + (size_t)layer * LAYERW,
                    Wi4 + (size_t)layer * LAYERW, mT, nT, lane, accr, acci);

  const int ln = lane & 15;
  const int hi = lane >> 4;
  const int col = nT * 16 + ln;          // output feature in [0,1024)
  const float biasr = br4[layer * 1024 + col];
  const float biasi = bi4[layer * 1024 + col];
  const int h = col >> 6;                // head
  const int d = col & 63;                // dim within head

#pragma unroll
  for (int r = 0; r < 8; ++r) {
    const int m = mT * 16 + r + 8 * hi;  // row in [0,4096)
    const int b = m >> 10;
    const int n = m & 1023;
    const size_t bh = (size_t)b * 16 + h;
    const __bf16 vr = (__bf16)(accr[r] + biasr);
    const __bf16 vi = (__bf16)(acci[r] + biasi);
    if (layer == 0) {
      Qr[(bh * 1024 + n) * 64 + d] = vr;
      Qi[(bh * 1024 + n) * 64 + d] = vi;
    } else if (layer == 1) {
      Kr[(bh * 1024 + n) * 64 + d] = vr;
      Ki[(bh * 1024 + n) * 64 + d] = vi;
    } else {
      Vrt[(bh * 64 + d) * 1024 + n] = vr;   // transposed for PV B-operand
      Vit[(bh * 64 + d) * 1024 + n] = vi;
    }
  }
}

// ---------------------------------------------------------------------------
// Flash-style complex-magnitude attention.
// One wave handles (bh, 16-query tile); 64*64 = 4096 waves, 4 waves/block.
// Per 32-key chunk: 16 WMMAs (sim_r/sim_i), magnitude+online softmax,
// P -> LDS -> A-operand, 8 WMMAs (P@Vr, P@Vi over 4 dh-tiles).
// ---------------------------------------------------------------------------
__global__ __launch_bounds__(128) void attn_flash_kernel(
    const __bf16* __restrict__ Qr, const __bf16* __restrict__ Qi,
    const __bf16* __restrict__ Kr, const __bf16* __restrict__ Ki,
    const __bf16* __restrict__ Vrt, const __bf16* __restrict__ Vit,
    __bf16* __restrict__ Or, __bf16* __restrict__ Oi) {
  __shared__ __align__(16) __bf16 Pb[4][16 * 32];

  const int warp = threadIdx.x >> 5;
  const int lane = threadIdx.x & 31;
  const int task = blockIdx.x * 4 + warp;
  const int bh = task >> 6;        // b*16 + h
  const int iT = task & 63;        // query tile
  const int ln = lane & 15;
  const int hi = lane >> 4;
  __bf16* Pw = Pb[warp];

  // Load the query tile once (A-operand layout, dh = 64 -> two K-chunks).
  const __bf16* qr_row = Qr + ((size_t)bh * 1024 + iT * 16 + ln) * 64 + hi * 8;
  const __bf16* qi_row = Qi + ((size_t)bh * 1024 + iT * 16 + ln) * 64 + hi * 8;
  const v16bf aqr0 = ld_seg2(qr_row);
  const v16bf aqr1 = ld_seg2(qr_row + 32);
  const v16bf aqi0 = ld_seg2(qi_row);
  const v16bf aqi1 = ld_seg2(qi_row + 32);
  const v16bf aqrn0 = negv(aqr0);
  const v16bf aqrn1 = negv(aqr1);

  v8f or_[4], oi_[4];
#pragma unroll
  for (int dt = 0; dt < 4; ++dt) { or_[dt] = vzero8(); oi_[dt] = vzero8(); }
  float m_[8], l_[8];
#pragma unroll
  for (int r = 0; r < 8; ++r) { m_[r] = -1.0e30f; l_[r] = 0.0f; }

  const float scale = 0.125f;  // 1/sqrt(DH=64)

  for (int jc = 0; jc < 1024; jc += 32) {
    v8f sr[2], si[2];
#pragma unroll
    for (int t = 0; t < 2; ++t) {
      const size_t krow = ((size_t)bh * 1024 + jc + t * 16 + ln) * 64 + hi * 8;
      const v16bf bkr0 = ld_cont16(Kr + krow);
      const v16bf bkr1 = ld_cont16(Kr + krow + 32);
      const v16bf bki0 = ld_cont16(Ki + krow);
      const v16bf bki1 = ld_cont16(Ki + krow + 32);
      v8f s = vzero8();                       // sim_r = Qr.Kr + Qi.Ki
      s = wmma_bf16(aqr0, bkr0, s);
      s = wmma_bf16(aqr1, bkr1, s);
      s = wmma_bf16(aqi0, bki0, s);
      s = wmma_bf16(aqi1, bki1, s);
      sr[t] = s;
      v8f u = vzero8();                       // sim_i = Qi.Kr - Qr.Ki
      u = wmma_bf16(aqi0, bkr0, u);
      u = wmma_bf16(aqi1, bkr1, u);
      u = wmma_bf16(aqrn0, bki0, u);
      u = wmma_bf16(aqrn1, bki1, u);
      si[t] = u;
    }

    // score = sqrt(sim_r^2 + sim_i^2) * scale, then online softmax update.
    float alpha[8];
#pragma unroll
    for (int r = 0; r < 8; ++r) {
      const float g0 = sqrtf(sr[0][r] * sr[0][r] + si[0][r] * si[0][r]) * scale;
      const float g1 = sqrtf(sr[1][r] * sr[1][r] + si[1][r] * si[1][r]) * scale;
      float c = fmaxf(g0, g1);                 // row-max over 32 keys
      c = fmaxf(c, __shfl_xor(c, 1));
      c = fmaxf(c, __shfl_xor(c, 2));
      c = fmaxf(c, __shfl_xor(c, 4));
      c = fmaxf(c, __shfl_xor(c, 8));
      const float mn = fmaxf(m_[r], c);
      const float al = expf(m_[r] - mn);
      const float p0 = expf(g0 - mn);
      const float p1 = expf(g1 - mn);
      float ps = p0 + p1;                      // row-sum over 32 keys
      ps += __shfl_xor(ps, 1);
      ps += __shfl_xor(ps, 2);
      ps += __shfl_xor(ps, 4);
      ps += __shfl_xor(ps, 8);
      l_[r] = l_[r] * al + ps;
      m_[r] = mn;
      alpha[r] = al;
      const int row = r + 8 * hi;              // P tile: [row][32 keys]
      Pw[row * 32 + ln] = (__bf16)p0;
      Pw[row * 32 + 16 + ln] = (__bf16)p1;
    }
#pragma unroll
    for (int dt = 0; dt < 4; ++dt)
#pragma unroll
      for (int r = 0; r < 8; ++r) {
        or_[dt][r] *= alpha[r];
        oi_[dt][r] *= alpha[r];
      }
    __syncthreads();

    // Reload P in A-operand layout and accumulate P@Vr / P@Vi.
    const v16bf ap = ld_seg2(Pw + ln * 32 + hi * 8);
#pragma unroll
    for (int dt = 0; dt < 4; ++dt) {
      const size_t vrow = ((size_t)bh * 64 + dt * 16 + ln) * 1024 + jc + hi * 8;
      const v16bf bvr = ld_cont16(Vrt + vrow);
      const v16bf bvi = ld_cont16(Vit + vrow);
      or_[dt] = wmma_bf16(ap, bvr, or_[dt]);
      oi_[dt] = wmma_bf16(ap, bvi, oi_[dt]);
    }
    __syncthreads();
  }

  // Normalize and write merged-head bf16 output [B*N, 1024].
  const int b = bh >> 4;
  const int h = bh & 15;
  float inv_l[8];
#pragma unroll
  for (int r = 0; r < 8; ++r) inv_l[r] = 1.0f / l_[r];
#pragma unroll
  for (int dt = 0; dt < 4; ++dt) {
    const int col = h * 64 + dt * 16 + ln;
#pragma unroll
    for (int r = 0; r < 8; ++r) {
      const int i = iT * 16 + r + 8 * hi;
      const size_t idx = ((size_t)b * 1024 + i) * 1024 + col;
      Or[idx] = (__bf16)(or_[dt][r] * inv_l[r]);
      Oi[idx] = (__bf16)(oi_[dt][r] * inv_l[r]);
    }
  }
}

// ---------------------------------------------------------------------------
// Output projection (layer 3), fp32 result straight into d_out [2,B,N,D].
// 256 mTiles x 64 nTiles = 16384 wave-tiles, 8 waves/block -> 2048 blocks.
// ---------------------------------------------------------------------------
__global__ __launch_bounds__(256) void proj_out_kernel(
    const __bf16* __restrict__ Xr, const __bf16* __restrict__ Xi,
    const __bf16* __restrict__ Wr, const __bf16* __restrict__ Wi,
    const float* __restrict__ br, const float* __restrict__ bi,
    float* __restrict__ out) {
  const int wave = blockIdx.x * 8 + (threadIdx.x >> 5);
  const int lane = threadIdx.x & 31;
  const int mT = wave >> 6;
  const int nT = wave & 63;

  v8f accr, acci;
  complex_gemm_tile(Xr, Xi, Wr, Wi, mT, nT, lane, accr, acci);

  const int ln = lane & 15;
  const int hi = lane >> 4;
  const int col = nT * 16 + ln;
  const float biasr = br[col];
  const float biasi = bi[col];
#pragma unroll
  for (int r = 0; r < 8; ++r) {
    const int m = mT * 16 + r + 8 * hi;        // row == b*1024 + n
    const size_t idx = (size_t)m * 1024 + col;
    out[idx] = accr[r] + biasr;                // real plane
    out[(size_t)BUFSZ + idx] = acci[r] + biasi;  // imag plane (+B*N*D)
  }
}

// ---------------------------------------------------------------------------
// Launcher. Inputs: xr, xi, Wr[4], Wi[4], br[4], bi[4] (all fp32).
// Workspace: 12 bf16 buffers x 4M elements = 96 MB.
// ---------------------------------------------------------------------------
extern "C" void kernel_launch(void* const* d_in, const int* in_sizes, int n_in,
                              void* d_out, int out_size, void* d_ws,
                              size_t ws_size, hipStream_t stream) {
  const float* xr = (const float*)d_in[0];
  const float* xi = (const float*)d_in[1];
  const float* Wr = (const float*)d_in[2];
  const float* Wi = (const float*)d_in[3];
  const float* br = (const float*)d_in[4];
  const float* bi = (const float*)d_in[5];

  __bf16* ws = (__bf16*)d_ws;
  __bf16* xr_bf = ws + 0 * (size_t)BUFSZ;
  __bf16* xi_bf = ws + 1 * (size_t)BUFSZ;
  __bf16* wr_bf = ws + 2 * (size_t)BUFSZ;   // 4 layers of 1024x1024
  __bf16* wi_bf = ws + 3 * (size_t)BUFSZ;
  __bf16* qr    = ws + 4 * (size_t)BUFSZ;   // [B,H,N,dh]
  __bf16* qi    = ws + 5 * (size_t)BUFSZ;
  __bf16* kr    = ws + 6 * (size_t)BUFSZ;
  __bf16* ki    = ws + 7 * (size_t)BUFSZ;
  __bf16* vrt   = ws + 8 * (size_t)BUFSZ;   // [B,H,dh,N] (transposed)
  __bf16* vit   = ws + 9 * (size_t)BUFSZ;
  __bf16* orb   = ws + 10 * (size_t)BUFSZ;  // attention out, merged heads
  __bf16* oib   = ws + 11 * (size_t)BUFSZ;

  // 1) fp32 -> bf16 conversions (each array is exactly 4M elements).
  convert_f32_bf16_kernel<<<2048, 256, 0, stream>>>(xr, xr_bf);
  convert_f32_bf16_kernel<<<2048, 256, 0, stream>>>(xi, xi_bf);
  convert_f32_bf16_kernel<<<2048, 256, 0, stream>>>(Wr, wr_bf);
  convert_f32_bf16_kernel<<<2048, 256, 0, stream>>>(Wi, wi_bf);

  // 2) Q/K/V complex projections (3 layers fused in one grid).
  proj_qkv_kernel<<<6144, 256, 0, stream>>>(xr_bf, xi_bf, wr_bf, wi_bf, br, bi,
                                            qr, qi, kr, ki, vrt, vit);

  // 3) Flash-style complex-magnitude attention.
  attn_flash_kernel<<<1024, 128, 0, stream>>>(qr, qi, kr, ki, vrt, vit, orb,
                                              oib);

  // 4) Output complex projection (layer 3) -> d_out fp32 [2,B,N,D].
  proj_out_kernel<<<2048, 256, 0, stream>>>(
      orb, oib, wr_bf + 3 * (size_t)LAYERW, wi_bf + 3 * (size_t)LAYERW,
      br + 3 * 1024, bi + 3 * 1024, (float*)d_out);
}
